// RecurrentRGCN_23759759082194
// MI455X (gfx1250) — matrix operand order repaired and checked by reading, not compiled, not run on hardware
//
#include <hip/hip_runtime.h>

namespace {

constexpr int N_ENT = 50000;
constexpr int H     = 256;
constexpr int R2    = 460;
constexpr int T_STEPS = 3;
constexpr int E_EDGE  = 100000;
constexpr int POS   = 3;
constexpr float SLOPE = (1.0f / 8.0f + 1.0f / 3.0f) * 0.5f;  // eval-mode rrelu

typedef __attribute__((ext_vector_type(16))) _Float16 v16h;
typedef __attribute__((ext_vector_type(8)))  float    v8f;

__device__ __forceinline__ float sigmoid_f(float x) { return 1.0f / (1.0f + expf(-x)); }

__device__ __forceinline__ float block_sum256(float v) {
  __shared__ float s[256];
  s[threadIdx.x] = v;
  __syncthreads();
  #pragma unroll
  for (int off = 128; off > 0; off >>= 1) {
    if ((int)threadIdx.x < off) s[threadIdx.x] += s[threadIdx.x + off];
    __syncthreads();
  }
  float r = s[0];
  __syncthreads();
  return r;
}

__device__ __forceinline__ void put4(v16h& f, int base, float4 q) {
  f[base + 0] = (_Float16)q.x;
  f[base + 1] = (_Float16)q.y;
  f[base + 2] = (_Float16)q.z;
  f[base + 3] = (_Float16)q.w;
}

// ----- A fragment 16x32 (MxK).  Lane half hl: two contiguous 8-float runs:
//   f[0..7]  <- Arow[k0 + hl*8 .. +7]
//   f[8..15] <- Arow[k0 + 16 + hl*8 .. +7]
__device__ __forceinline__ v16h load_a16(const float* __restrict__ Arow, int k0, int hl) {
  const float* p0 = Arow + k0 + hl * 8;
  const float* p1 = Arow + k0 + 16 + hl * 8;
  v16h f;
  put4(f, 0,  *(const float4*)(p0));
  put4(f, 4,  *(const float4*)(p0 + 4));
  put4(f, 8,  *(const float4*)(p1));
  put4(f, 12, *(const float4*)(p1 + 4));
  return f;
}

// A fragment where the row is the elementwise sum of two rows (msg GEMM).
__device__ __forceinline__ float4 add4(float4 a, float4 b) {
  float4 c; c.x = a.x + b.x; c.y = a.y + b.y; c.z = a.z + b.z; c.w = a.w + b.w;
  return c;
}
__device__ __forceinline__ v16h load_a16_sum(const float* __restrict__ r0,
                                             const float* __restrict__ r1,
                                             int k0, int hl) {
  const int o0 = k0 + hl * 8, o1 = k0 + 16 + hl * 8;
  v16h f;
  put4(f, 0,  add4(*(const float4*)(r0 + o0),     *(const float4*)(r1 + o0)));
  put4(f, 4,  add4(*(const float4*)(r0 + o0 + 4), *(const float4*)(r1 + o0 + 4)));
  put4(f, 8,  add4(*(const float4*)(r0 + o1),     *(const float4*)(r1 + o1)));
  put4(f, 12, add4(*(const float4*)(r0 + o1 + 4), *(const float4*)(r1 + o1 + 4)));
  return f;
}

// ----- B fragment 32x16 (KxN), B stored row-per-output-col (B^T layout):
//   f[0..7]  <- Brow[k0 + hl*8 .. +7],  f[8..15] <- Brow[k0 + 16 + hl*8 .. +7]
__device__ __forceinline__ v16h load_b16(const float* __restrict__ Brow, int k0, int hl) {
  const float* p = Brow + k0 + hl * 8;
  const float* q = Brow + k0 + 16 + hl * 8;
  v16h f;
  put4(f, 0,  *(const float4*)(p));
  put4(f, 4,  *(const float4*)(p + 4));
  put4(f, 8,  *(const float4*)(q));
  put4(f, 12, *(const float4*)(q + 4));
  return f;
}

// ---------------- Generic GEMM: C = act( A[M,K] x Bt[Nn,K]^T + bias ) -------
// Requires K % 32 == 0, lda/ldb multiples of 4. act: 0=none,1=tanh,2=sigmoid.
// 256 threads = 8 waves -> 4x2 tiles of 16x16 -> 64x32 per block.
__global__ __launch_bounds__(256)
void wmma_gemm_kernel(const float* __restrict__ A, int lda,
                      const float* __restrict__ Bt, int ldb,
                      const float* __restrict__ bias, int act,
                      float* __restrict__ C, int ldc,
                      int M, int Nn, int K) {
  const int lane = threadIdx.x & 31;
  const int wave = threadIdx.x >> 5;
  const int row0 = blockIdx.y * 64 + (wave & 3) * 16;
  const int col0 = blockIdx.x * 32 + (wave >> 2) * 16;
  if (row0 >= M) return;  // wave-uniform exit

  const int hl  = lane >> 4;
  const int m_c = min(row0 + (lane & 15), M - 1);   // clamp: edge rows recomputed, not stored
  const int n_c = min(col0 + (lane & 15), Nn - 1);
  const float* __restrict__ Arow = A  + (size_t)m_c * lda;
  const float* __restrict__ Brow = Bt + (size_t)n_c * ldb;

  v8f acc = {};
  for (int k0 = 0; k0 < K; k0 += 32) {
    v16h a = load_a16(Arow, k0, hl);
    v16h b = load_b16(Brow, k0, hl);
    acc = __builtin_amdgcn_wmma_f32_16x16x32_f16(false, a, false, b,
                                                 (short)0, acc, false, false);
  }

  const int col   = col0 + (lane & 15);
  const int rbase = row0 + hl * 8;
  const float bv  = bias ? bias[min(col, Nn - 1)] : 0.0f;
  #pragma unroll
  for (int i = 0; i < 8; ++i) {
    const int r = rbase + i;
    if (r < M && col < Nn) {
      float v = acc[i] + bv;
      if (act == 1)      v = tanhf(v);
      else if (act == 2) v = sigmoid_f(v);
      C[(size_t)r * ldc + col] = v;
    }
  }
}

// ----- Attention GEMM: C = tanh([nh | pos] @ W_hp.T + b_hp), K = 256 + 3 ----
// posPad: [M,32] zero-padded pos rows; WhpP: [256,260] padded W_hp (+64 slack).
__global__ __launch_bounds__(256)
void atten_gemm_kernel(const float* __restrict__ nh, const float* __restrict__ posPad,
                       const float* __restrict__ WhpP, const float* __restrict__ bias,
                       float* __restrict__ C, int M) {
  const int lane = threadIdx.x & 31;
  const int wave = threadIdx.x >> 5;
  const int row0 = blockIdx.y * 64 + (wave & 3) * 16;
  const int col0 = blockIdx.x * 32 + (wave >> 2) * 16;
  if (row0 >= M) return;

  const int hl  = lane >> 4;
  const int m_c = min(row0 + (lane & 15), M - 1);
  const float* __restrict__ Arow = nh     + (size_t)m_c * H;
  const float* __restrict__ Prow = posPad + (size_t)m_c * 32;
  const float* __restrict__ Brow = WhpP   + (size_t)(col0 + (lane & 15)) * 260;

  v8f acc = {};
  for (int k0 = 0; k0 < H; k0 += 32) {
    v16h a = load_a16(Arow, k0, hl);
    v16h b = load_b16(Brow, k0, hl);
    acc = __builtin_amdgcn_wmma_f32_16x16x32_f16(false, a, false, b,
                                                 (short)0, acc, false, false);
  }
  // K-tail (3 valid cols): A side zero-padded to 32, so overrun B products vanish.
  {
    v16h a = load_a16(Prow, 0, hl);
    v16h b = load_b16(Brow, 256, hl);
    acc = __builtin_amdgcn_wmma_f32_16x16x32_f16(false, a, false, b,
                                                 (short)0, acc, false, false);
  }

  const int col   = col0 + (lane & 15);
  const int rbase = row0 + hl * 8;
  const float bv  = bias[col];
  #pragma unroll
  for (int i = 0; i < 8; ++i) {
    const int r = rbase + i;
    if (r < M) C[(size_t)r * H + col] = tanhf(acc[i] + bv);
  }
}

// ----- Message GEMM fused with scatter-add:
//   agg[dst[e]] += (nh[src[e]] + h0[etype[e]]) @ W_neigh   (WnT pre-transposed)
__global__ __launch_bounds__(256)
void msg_gemm_scatter_kernel(const float* __restrict__ nh, const float* __restrict__ h0,
                             const int* __restrict__ src, const int* __restrict__ ety,
                             const int* __restrict__ dst,
                             const float* __restrict__ WnT, float* __restrict__ agg, int E) {
  const int lane = threadIdx.x & 31;
  const int wave = threadIdx.x >> 5;
  const int row0 = blockIdx.y * 64 + (wave & 3) * 16;
  const int col0 = blockIdx.x * 32 + (wave >> 2) * 16;
  if (row0 >= E) return;

  const int hl  = lane >> 4;
  const int m_c = min(row0 + (lane & 15), E - 1);
  const int s = src[m_c];
  const int e = ety[m_c];
  const float* __restrict__ nrow = nh + (size_t)s * H;
  const float* __restrict__ rrow = h0 + (size_t)e * H;
  const float* __restrict__ Brow = WnT + (size_t)(col0 + (lane & 15)) * H;

  v8f acc = {};
  for (int k0 = 0; k0 < H; k0 += 32) {
    v16h a = load_a16_sum(nrow, rrow, k0, hl);
    v16h b = load_b16(Brow, k0, hl);
    acc = __builtin_amdgcn_wmma_f32_16x16x32_f16(false, a, false, b,
                                                 (short)0, acc, false, false);
  }

  const int col   = col0 + (lane & 15);
  const int rbase = row0 + hl * 8;
  #pragma unroll
  for (int i = 0; i < 8; ++i) {
    const int er = rbase + i;
    if (er < E) {
      const int d = dst[er];
      atomicAdd(&agg[(size_t)d * H + col], acc[i]);
    }
  }
}

// ----- Self-loop GEMM (both weights, pre-transposed) + layer epilogue -------
// nh_out = leaky( agg*norm + (deg>0 ? nh@W_loop : nh@W_evolve) )
__global__ __launch_bounds__(256)
void loop_gemm_layer_kernel(const float* __restrict__ nh,
                            const float* __restrict__ WlT, const float* __restrict__ WeT,
                            const float* __restrict__ agg, const float* __restrict__ deg,
                            float* __restrict__ nh_out, int M) {
  const int lane = threadIdx.x & 31;
  const int wave = threadIdx.x >> 5;
  const int row0 = blockIdx.y * 64 + (wave & 3) * 16;
  const int col0 = blockIdx.x * 32 + (wave >> 2) * 16;
  if (row0 >= M) return;

  const int hl  = lane >> 4;
  const int m_c = min(row0 + (lane & 15), M - 1);
  const float* __restrict__ Arow  = nh  + (size_t)m_c * H;
  const float* __restrict__ Brow1 = WlT + (size_t)(col0 + (lane & 15)) * H;
  const float* __restrict__ Brow2 = WeT + (size_t)(col0 + (lane & 15)) * H;

  v8f acc1 = {}, acc2 = {};
  for (int k0 = 0; k0 < H; k0 += 32) {
    v16h a  = load_a16(Arow, k0, hl);
    v16h b1 = load_b16(Brow1, k0, hl);
    v16h b2 = load_b16(Brow2, k0, hl);
    acc1 = __builtin_amdgcn_wmma_f32_16x16x32_f16(false, a, false, b1,
                                                  (short)0, acc1, false, false);
    acc2 = __builtin_amdgcn_wmma_f32_16x16x32_f16(false, a, false, b2,
                                                  (short)0, acc2, false, false);
  }

  const int col   = col0 + (lane & 15);
  const int rbase = row0 + hl * 8;
  #pragma unroll
  for (int i = 0; i < 8; ++i) {
    const int r = rbase + i;
    if (r < M) {
      const float d   = deg[r];
      const float nrm = (d > 0.0f) ? (1.0f / fmaxf(d, 1.0f)) : 0.0f;
      const float lp  = (d > 0.0f) ? acc1[i] : acc2[i];
      float pre = agg[(size_t)r * H + col] * nrm + lp;
      nh_out[(size_t)r * H + col] = (pre >= 0.0f) ? pre : SLOPE * pre;
    }
  }
}

// ---------------- preprocessing kernels -------------------------------------
// out[n*256 + k] = in[k*256 + n]   (LDS-tiled 256x256 transpose)
__global__ __launch_bounds__(256)
void transpose256_kernel(const float* __restrict__ in, float* __restrict__ out) {
  __shared__ float tile[16][17];
  const int bx = blockIdx.x * 16, by = blockIdx.y * 16;
  const int tx = threadIdx.x & 15, ty = threadIdx.x >> 4;
  tile[ty][tx] = in[(size_t)(by + ty) * 256 + bx + tx];
  __syncthreads();
  out[(size_t)(bx + ty) * 256 + by + tx] = tile[tx][ty];
}

// W_hp [256,259] -> padded [256,260] (col 259 = 0)
__global__ __launch_bounds__(288)
void pad_whp_kernel(const float* __restrict__ in, float* __restrict__ out) {
  const int r = blockIdx.x, j = threadIdx.x;
  if (j < 260) out[(size_t)r * 260 + j] = (j < 259) ? in[(size_t)r * 259 + j] : 0.0f;
}

// pos [M,3] -> padded [M,32]
__global__ __launch_bounds__(256)
void pad_pos_kernel(const float* __restrict__ pos, float* __restrict__ out, int M) {
  const int idx = blockIdx.x * 256 + threadIdx.x;
  const int m = idx >> 5, j = idx & 31;
  if (m < M) out[(size_t)m * 32 + j] = (j < POS) ? pos[(size_t)m * POS + j] : 0.0f;
}

// ---------------- elementwise / scatter kernels -----------------------------
__global__ __launch_bounds__(256)
void l2norm_rows_kernel(const float* __restrict__ in, float* __restrict__ out) {
  const size_t i = (size_t)blockIdx.x * H + threadIdx.x;
  const float v = in[i];
  const float ss = block_sum256(v * v);
  out[i] = v / fmaxf(sqrtf(ss), 1e-12f);
}

__global__ __launch_bounds__(256)
void seg_accum_kernel(const float* __restrict__ h, const int* __restrict__ r_to_e,
                      const int* __restrict__ r_seg,
                      float* __restrict__ sums, float* __restrict__ cnt) {
  const int e   = blockIdx.x;
  const int row = r_to_e[e];
  const int seg = r_seg[e];
  atomicAdd(&sums[(size_t)seg * H + threadIdx.x], h[(size_t)row * H + threadIdx.x]);
  if (threadIdx.x == 0) atomicAdd(&cnt[seg], 1.0f);
}

__global__ __launch_bounds__(256)
void build_xin_kernel(const float* __restrict__ emb_rel, const float* __restrict__ sums,
                      const float* __restrict__ cnt, float* __restrict__ x_in) {
  const int r = blockIdx.x, j = threadIdx.x;
  const float c = cnt[r];
  x_in[(size_t)r * 2 * H + j]     = emb_rel[(size_t)r * H + j];
  x_in[(size_t)r * 2 * H + H + j] =
      (c > 0.0f) ? sums[(size_t)r * H + j] / fmaxf(c, 1.0f) : 0.0f;
}

__global__ __launch_bounds__(256)
void gru_update_kernel(const float* __restrict__ gi, const float* __restrict__ gh,
                       float* __restrict__ h0) {
  const int r = blockIdx.x, j = threadIdx.x;
  const float* gir = gi + (size_t)r * 3 * H;
  const float* ghr = gh + (size_t)r * 3 * H;
  const float rr = sigmoid_f(gir[j] + ghr[j]);
  const float zz = sigmoid_f(gir[H + j] + ghr[H + j]);
  const float nn = tanhf(gir[2 * H + j] + rr * ghr[2 * H + j]);
  const float hp = h0[(size_t)r * H + j];
  const float v  = (1.0f - zz) * nn + zz * hp;
  const float ss = block_sum256(v * v);
  h0[(size_t)r * H + j] = v / fmaxf(sqrtf(ss), 1e-12f);
}

__global__ __launch_bounds__(256)
void deg_count_kernel(const int* __restrict__ dst, float* __restrict__ deg, int E) {
  const int e = blockIdx.x * blockDim.x + threadIdx.x;
  if (e < E) atomicAdd(&deg[dst[e]], 1.0f);
}

__global__ __launch_bounds__(256)
void time_combine_kernel(const float* __restrict__ nh, const float* __restrict__ tw,
                         float* __restrict__ h) {
  const size_t i = (size_t)blockIdx.x * H + threadIdx.x;
  const float v  = nh[i];
  const float ss = block_sum256(v * v);
  const float cn = v / fmaxf(sqrtf(ss), 1e-12f);
  const float t  = tw[i];
  h[i] = t * cn + (1.0f - t) * h[i];
}

__global__ __launch_bounds__(256)
void out_combine_kernel(const float* __restrict__ gw, const float* __restrict__ gl,
                        const float* __restrict__ h, float* __restrict__ out) {
  const size_t i = (size_t)blockIdx.x * H + threadIdx.x;
  const float g = gw[i];
  out[i] = g * gl[i] + (1.0f - g) * h[i];
}

}  // namespace

extern "C" void kernel_launch(void* const* d_in, const int* in_sizes, int n_in,
                              void* d_out, int out_size, void* d_ws, size_t ws_size,
                              hipStream_t stream) {
  (void)in_sizes; (void)n_in; (void)out_size; (void)ws_size;
  const int*   src_all   = (const int*)d_in[0];
  const int*   dst_all   = (const int*)d_in[1];
  const int*   ety_all   = (const int*)d_in[2];
  const int*   r2e_all   = (const int*)d_in[3];
  const int*   rsg_all   = (const int*)d_in[4];
  const float* pos_enc   = (const float*)d_in[5];
  const float* dyn_emb   = (const float*)d_in[6];
  const float* emb_rel   = (const float*)d_in[7];
  const float* gW_ih     = (const float*)d_in[8];
  const float* gW_hh     = (const float*)d_in[9];
  const float* gb_ih     = (const float*)d_in[10];
  const float* gb_hh     = (const float*)d_in[11];
  const float* W_neigh   = (const float*)d_in[12];
  const float* W_loop    = (const float*)d_in[13];
  const float* W_evolve  = (const float*)d_in[14];
  const float* W_hp      = (const float*)d_in[15];
  const float* b_hp      = (const float*)d_in[16];
  const float* W_v       = (const float*)d_in[17];
  const float* b_v       = (const float*)d_in[18];
  const float* W_o       = (const float*)d_in[19];
  const float* b_o       = (const float*)d_in[20];
  const float* time_W    = (const float*)d_in[21];
  const float* time_b    = (const float*)d_in[22];
  const float* glob_W    = (const float*)d_in[23];
  const float* glob_b    = (const float*)d_in[24];
  float* out = (float*)d_out;

  const size_t NH  = (size_t)N_ENT * H;
  const size_t HH2 = (size_t)H * H;
  float* f     = (float*)d_ws;
  float* h     = f + 0 * NH;
  float* nh1   = f + 1 * NH;
  float* nh2   = f + 2 * NH;
  float* agg   = f + 3 * NH;  // reused as global_h after layers
  float* atten = f + 4 * NH;  // reused as gw at the end
  float* vtmp  = f + 5 * NH;
  float* tw    = f + 6 * NH;
  float* deg   = f + 7 * NH;                    // N floats
  float* sums  = deg  + N_ENT;                  // R2*H
  float* cnt   = sums + (size_t)R2 * H;         // R2
  float* x_in  = cnt  + R2;                     // R2*2H
  float* gi    = x_in + (size_t)R2 * 2 * H;     // R2*3H
  float* gh    = gi   + (size_t)R2 * 3 * H;     // R2*3H
  float* h0    = gh   + (size_t)R2 * 3 * H;     // R2*H
  float* WneighT = h0 + (size_t)R2 * H;         // 2*H*H
  float* WloopT  = WneighT + 2 * HH2;           // 2*H*H
  float* WevolveT= WloopT  + 2 * HH2;           // 2*H*H
  float* timeWT  = WevolveT + 2 * HH2;          // H*H
  float* globWT  = timeWT + HH2;                // H*H
  float* WhpP    = globWT + HH2;                // 256*260 + 64 slack
  float* posPad  = WhpP + (256 * 260 + 64);     // N*32

  auto gemm = [&](const float* A, int lda, const float* Bt, int ldb,
                  const float* bias, int act, float* C, int ldc,
                  int M, int Nn, int K) {
    dim3 g((Nn + 31) / 32, (M + 63) / 64);
    wmma_gemm_kernel<<<g, 256, 0, stream>>>(A, lda, Bt, ldb, bias, act, C, ldc, M, Nn, K);
  };

  // ---- preprocessing: transpose x@W weights, pad W_hp (all tiny vs 192MB L2)
  {
    dim3 tg(16, 16);
    for (int l = 0; l < 2; ++l) {
      transpose256_kernel<<<tg, 256, 0, stream>>>(W_neigh  + l * HH2, WneighT  + l * HH2);
      transpose256_kernel<<<tg, 256, 0, stream>>>(W_loop   + l * HH2, WloopT   + l * HH2);
      transpose256_kernel<<<tg, 256, 0, stream>>>(W_evolve + l * HH2, WevolveT + l * HH2);
    }
    transpose256_kernel<<<tg, 256, 0, stream>>>(time_W, timeWT);
    transpose256_kernel<<<tg, 256, 0, stream>>>(glob_W, globWT);
    pad_whp_kernel<<<256, 288, 0, stream>>>(W_hp, WhpP);
  }

  // h = l2norm(dynamic_emb);  h0 = emb_rel
  l2norm_rows_kernel<<<N_ENT, 256, 0, stream>>>(dyn_emb, h);
  hipMemcpyAsync(h0, emb_rel, (size_t)R2 * H * sizeof(float),
                 hipMemcpyDeviceToDevice, stream);

  const dim3 grid_msg(H / 32, (E_EDGE + 63) / 64);
  const dim3 grid_ent(H / 32, (N_ENT + 63) / 64);

  for (int t = 0; t < T_STEPS; ++t) {
    const int* srcT = src_all + (size_t)t * E_EDGE;
    const int* dstT = dst_all + (size_t)t * E_EDGE;
    const int* etyT = ety_all + (size_t)t * E_EDGE;
    const int* r2eT = r2e_all + (size_t)t * E_EDGE;
    const int* rsgT = rsg_all + (size_t)t * E_EDGE;
    const float* posT = pos_enc + (size_t)t * N_ENT * POS;

    // ---- per-relation mean + GRU -> h0
    hipMemsetAsync(sums, 0, (size_t)R2 * H * sizeof(float), stream);
    hipMemsetAsync(cnt, 0, (size_t)R2 * sizeof(float), stream);
    seg_accum_kernel<<<E_EDGE, 256, 0, stream>>>(h, r2eT, rsgT, sums, cnt);
    build_xin_kernel<<<R2, 256, 0, stream>>>(emb_rel, sums, cnt, x_in);
    gemm(x_in, 2 * H, gW_ih, 2 * H, gb_ih, 0, gi, 3 * H, R2, 3 * H, 2 * H);
    gemm(h0,   H,     gW_hh, H,     gb_hh, 0, gh, 3 * H, R2, 3 * H, H);
    gru_update_kernel<<<R2, 256, 0, stream>>>(gi, gh, h0);

    // ---- degree/norm for this snapshot
    hipMemsetAsync(deg, 0, (size_t)N_ENT * sizeof(float), stream);
    deg_count_kernel<<<(E_EDGE + 255) / 256, 256, 0, stream>>>(dstT, deg, E_EDGE);

    // ---- 2 RGCN layers (msg GEMM fused w/ scatter-add; loop GEMM fused w/ epilogue)
    const float* cur = h;
    float* nxt = nh1;
    for (int l = 0; l < 2; ++l) {
      hipMemsetAsync(agg, 0, NH * sizeof(float), stream);
      msg_gemm_scatter_kernel<<<grid_msg, 256, 0, stream>>>(
          cur, h0, srcT, etyT, dstT, WneighT + (size_t)l * HH2, agg, E_EDGE);
      loop_gemm_layer_kernel<<<grid_ent, 256, 0, stream>>>(
          cur, WloopT + (size_t)l * HH2, WevolveT + (size_t)l * HH2,
          agg, deg, nxt, N_ENT);
      cur = nxt;
      nxt = nh2;
    }

    // ---- attention head: tanh([nh|pos] @ W_hp.T + b), then v/o projections
    pad_pos_kernel<<<(N_ENT * 32 + 255) / 256, 256, 0, stream>>>(posT, posPad, N_ENT);
    atten_gemm_kernel<<<grid_ent, 256, 0, stream>>>(cur, posPad, WhpP, b_hp, atten, N_ENT);
    gemm(atten, H, W_v, H, b_v, 0, vtmp, H, N_ENT, H, H);
    gemm(vtmp,  H, W_o, H, b_o, 0, agg,  H, N_ENT, H, H);  // global_h

    // ---- time gate: h = tw * l2norm(nh) + (1-tw) * h
    gemm(h, H, timeWT, H, time_b, 2, tw, H, N_ENT, H, H);
    time_combine_kernel<<<N_ENT, 256, 0, stream>>>(cur, tw, h);

    // ---- global gate + output
    gemm(h, H, globWT, H, glob_b, 2, atten, H, N_ENT, H, H);  // gw
    out_combine_kernel<<<N_ENT, 256, 0, stream>>>(atten, agg, h, out + (size_t)t * NH);
  }
}
